// BranchingLayer_12549894439099
// MI455X (gfx1250) — matrix operand order repaired and checked by reading, not compiled, not run on hardware
//
#include <hip/hip_runtime.h>

typedef __attribute__((ext_vector_type(16))) _Float16 v16h;
typedef __attribute__((ext_vector_type(8)))  _Float16 v8h;
typedef __attribute__((ext_vector_type(8)))  float    v8f;
typedef __attribute__((ext_vector_type(4)))  float    v4f;

#define N_PARENTS 256
#define BATCH     1024
#define N_FEAT    128
#define N_GLOBAL  64
#define N_ROWS    (N_PARENTS * BATCH)   // 262144
#define D_IN      (N_FEAT + N_GLOBAL)   // 192
#define HID       256
#define KP1       (D_IN / 32)           // 6 k-panels for GEMM1
#define KP2       (HID / 32)            // 8 k-panels for GEMM2
#define NT        (HID / 16)            // 16 n-tiles
#define TILE_M    128                   // rows per workgroup (8 waves x 16)

#define W1_HALVES (KP1 * NT * 512)      // 49152 halves = 96 KB
#define W2_HALVES (KP2 * NT * 512)      // 65536 halves = 128 KB
#define W_HALVES  (W1_HALVES + W2_HALVES)
#define W_BYTES   (W_HALVES * 2)        // 229376 B
#define W1_CHUNKS (W1_HALVES * 2 / 16)  // 6144 x 16B
#define W_CHUNKS  (W_BYTES / 16)        // 14336 x 16B

// ---------------------------------------------------------------------------
// Pack W1 (192x256) and W2 (256x256) f32 -> f16 WMMA B-fragment layout.
// B 32x16 f16 fragment (wave32): lane l holds column n = nt*16 + (l&15);
// half j of the lane's v16h holds K = kp*32 + (j<8 ? j : j+8) + 8*(l>>4).
// ---------------------------------------------------------------------------
__global__ void prep_wfrag(const float* __restrict__ W1,
                           const float* __restrict__ W2,
                           _Float16* __restrict__ wf) {
    int t    = blockIdx.x * blockDim.x + threadIdx.x;
    int lane = t & 31;
    int frag = t >> 5;
    const int NW1 = KP1 * NT;                 // 96 fragments
    const int NW2 = KP2 * NT;                 // 128 fragments
    if (frag >= NW1 + NW2) return;

    const float* W;
    int kp, nt;
    if (frag < NW1) { W = W1; kp = frag / NT; nt = frag % NT; }
    else            { W = W2; kp = (frag - NW1) / NT; nt = (frag - NW1) % NT; }

    int n     = nt * 16 + (lane & 15);
    int kbase = kp * 32 + 8 * (lane >> 4);
    _Float16* o = wf + ((size_t)frag * 32 + lane) * 16;
#pragma unroll
    for (int j = 0; j < 16; ++j)
        o[j] = (_Float16)W[(size_t)(kbase + (j < 8 ? j : j + 8)) * HID + n];
}

// ---------------------------------------------------------------------------
// Copy x into the head of the output (the concat([x, children]) part).
// ---------------------------------------------------------------------------
__global__ void copy_x(const float4* __restrict__ src, float4* __restrict__ dst, int n4) {
    int i = blockIdx.x * blockDim.x + threadIdx.x;
    if (i < n4) dst[i] = src[i];
}

// ---------------------------------------------------------------------------
// Fused FFN: async-stage packed weights into LDS, then
// gather -> GEMM1(+bias,LeakyReLU) -> GEMM2(+bias,residual) -> scatter.
// ---------------------------------------------------------------------------
__global__ __launch_bounds__(256)
void ffn_kernel(const float* __restrict__ x,
                const float* __restrict__ gf,
                const float* __restrict__ b1,
                const float* __restrict__ b2,
                const _Float16* __restrict__ wfrag,
                const int* __restrict__ idxs_level,
                const int* __restrict__ parents_idxs,
                float* __restrict__ out) {
    __shared__ __align__(16) _Float16 Wls[W_HALVES];         // 224 KB packed weights
    __shared__ __align__(16) _Float16 Hs[TILE_M][HID + 8];   // 66 KB f16 hidden

    const int tid  = threadIdx.x;
    const int lane = tid & 31;
    const int w    = tid >> 5;          // wave id 0..7 -> 16-row m-tile
    const int lm   = lane & 15;
    const int g    = lane >> 4;
    const int rowBase = blockIdx.x * TILE_M + w * 16;

    // ---- async-stage weight fragments global->LDS (ASYNCcnt tracked) ------
    // chunk c = tid + j*256: consecutive lanes hit consecutive 16B -> each
    // instruction moves a coalesced 512B per wave.  W1 chunks first, then W2,
    // so `s_wait_asynccnt <= 32` (= W2 chunks/thread) guarantees W1 landed
    // while W2 staging still overlaps GEMM1 compute.
    {
        const uint32_t ldsBase = (uint32_t)(uintptr_t)&Wls[0];
        const char*    gsrc    = (const char*)wfrag;
#pragma unroll 1
        for (int j = 0; j < W_CHUNKS / 256; ++j) {          // 56 chunks/thread
            const int      c = tid + j * 256;
            const uint64_t ga = (uint64_t)(uintptr_t)(gsrc + (size_t)c * 16);
            const uint32_t la = ldsBase + (uint32_t)c * 16;
            asm volatile("global_load_async_to_lds_b128 %0, %1, off"
                         :: "v"(la), "v"(ga) : "memory");
        }
    }

    // ---- A-fragments for GEMM1 (16-bit A 16x32 layout: K=(j<8?j:j+8)+8g) ----
    const int arow = rowBase + lm;                    // lanes 0-15 & 16-31: same rows
    const int srow = idxs_level[arow];                // honor the gather
    const float* xr = x  + (size_t)srow * N_FEAT;
    const int gidx  = parents_idxs[arow] & (BATCH - 1);
    const float* gr = gf + (size_t)gidx * N_GLOBAL;

    v16h a1[KP1];
#pragma unroll
    for (int kp = 0; kp < KP1; ++kp) {
        const int k0 = kp * 32 + 8 * g;               // run0 = k0..k0+7, run1 = k0+16..k0+23
        const float* base = (kp < 4) ? xr : gr;
        const int off     = (kp < 4) ? k0 : (k0 - N_FEAT);
        v4f u0 = *(const v4f*)(base + off);
        v4f u1 = *(const v4f*)(base + off + 4);
        v4f u2 = *(const v4f*)(base + off + 16);
        v4f u3 = *(const v4f*)(base + off + 20);
        v16h a;
#pragma unroll
        for (int j = 0; j < 4; ++j) {
            a[j]      = (_Float16)u0[j];
            a[4 + j]  = (_Float16)u1[j];
            a[8 + j]  = (_Float16)u2[j];
            a[12 + j] = (_Float16)u3[j];
        }
        a1[kp] = a;
    }

    // wait for W1 fragments only (W2 still in flight), then make visible
    asm volatile("s_wait_asynccnt 0x20" ::: "memory");  // <= 32 outstanding
    __syncthreads();

    // ---- GEMM1: H = leaky_relu([x|g] @ W1 + b1), f32 accum, f16 to LDS ----
#pragma unroll 1
    for (int nt = 0; nt < NT; ++nt) {
        const int c = nt * 16 + lm;
        const float bias = b1[c];
        v8f acc;
#pragma unroll
        for (int e = 0; e < 8; ++e) acc[e] = bias;
#pragma unroll
        for (int kp = 0; kp < KP1; ++kp) {
            const v8h* bp = (const v8h*)&Wls[((kp * NT + nt) * 32 + lane) * 16];
            v8h bl = bp[0], bh = bp[1];
            v16h bf;
#pragma unroll
            for (int j = 0; j < 8; ++j) { bf[j] = bl[j]; bf[8 + j] = bh[j]; }
            acc = __builtin_amdgcn_wmma_f32_16x16x32_f16(
                false, a1[kp], false, bf, (short)0, acc, false, false);
        }
#pragma unroll
        for (int e = 0; e < 8; ++e) {
            float v = acc[e];
            v = fmaxf(v, 0.01f * v);                   // leaky_relu: slope<1 => max(x,0.01x)
            Hs[w * 16 + e + 8 * g][c] = (_Float16)v;   // D layout: row = e+8g, col = c
        }
    }

    // wait for W2 fragments, make W2 + Hs visible
    asm volatile("s_wait_asynccnt 0x0" ::: "memory");
    __syncthreads();

    // ---- A-fragments for GEMM2 from LDS H (16B-aligned ds loads) ----
    v16h a2[KP2];
#pragma unroll
    for (int kp = 0; kp < KP2; ++kp) {
        const int k0 = kp * 32 + 8 * g;
        v8h u0 = *(const v8h*)&Hs[w * 16 + lm][k0];
        v8h u1 = *(const v8h*)&Hs[w * 16 + lm][k0 + 16];
        v16h a;
#pragma unroll
        for (int j = 0; j < 8; ++j) { a[j] = u0[j]; a[8 + j] = u1[j]; }
        a2[kp] = a;
    }

    // ---- per-row scatter bases: outRow = N_ROWS + 2p*1024 + b (+ br*1024) ----
    size_t obase[8];
    const float* xres[8];
#pragma unroll
    for (int e = 0; e < 8; ++e) {
        const int rg = rowBase + e + 8 * g;           // D-tile row for acc element e
        xres[e] = x + (size_t)idxs_level[rg] * N_FEAT;
        const int p  = rg >> 10;
        const int bb = rg & (BATCH - 1);
        obase[e] = ((size_t)N_ROWS + ((size_t)p << 11) + (size_t)bb) * N_FEAT;
    }

    // ---- GEMM2 + bias + interleaved residual + scatter ----
    const _Float16* w2l = &Wls[W1_HALVES];
#pragma unroll 1
    for (int nt = 0; nt < NT; ++nt) {
        const int c = nt * 16 + lm;
        const float bias = b2[c];
        v8f acc;
#pragma unroll
        for (int e = 0; e < 8; ++e) acc[e] = bias;
#pragma unroll
        for (int kp = 0; kp < KP2; ++kp) {
            const v8h* bp = (const v8h*)(w2l + ((size_t)(kp * NT + nt) * 32 + lane) * 16);
            v8h bl = bp[0], bh = bp[1];
            v16h bf;
#pragma unroll
            for (int j = 0; j < 8; ++j) { bf[j] = bl[j]; bf[8 + j] = bh[j]; }
            acc = __builtin_amdgcn_wmma_f32_16x16x32_f16(
                false, a2[kp], false, bf, (short)0, acc, false, false);
        }
        const int    br    = c >> 7;                  // branch = c/128
        const int    f     = c & (N_FEAT - 1);        // child feature = c%128
        const size_t broff = (size_t)br * ((size_t)BATCH * N_FEAT) + f;
        const int    half  = c >> 1;                  // repeat(x,2): col c <- x[:, c>>1]
#pragma unroll
        for (int e = 0; e < 8; ++e)
            out[obase[e] + broff] = acc[e] + xres[e][half];
    }
}

// ---------------------------------------------------------------------------
extern "C" void kernel_launch(void* const* d_in, const int* in_sizes, int n_in,
                              void* d_out, int out_size, void* d_ws, size_t ws_size,
                              hipStream_t stream) {
    const float* x    = (const float*)d_in[0];
    const float* gf   = (const float*)d_in[1];
    const float* W1   = (const float*)d_in[2];
    const float* b1   = (const float*)d_in[3];
    const float* W2   = (const float*)d_in[4];
    const float* b2   = (const float*)d_in[5];
    const int*   idxs = (const int*)d_in[6];
    const int*   pidx = (const int*)d_in[7];
    float*       out  = (float*)d_out;
    _Float16*    wfrag = (_Float16*)d_ws;            // 224 KB of scratch used

    // 1) pack weights into f16 WMMA B-fragments (L2-resident afterwards)
    const int nPackThreads = (KP1 + KP2) * NT * 32;  // 7168
    prep_wfrag<<<(nPackThreads + 255) / 256, 256, 0, stream>>>(W1, W2, wfrag);

    // 2) concat head: out[0 : N_ROWS*128] = x
    const int n4 = N_ROWS * N_FEAT / 4;
    copy_x<<<(n4 + 255) / 256, 256, 0, stream>>>((const float4*)x, (float4*)out, n4);

    // 3) fused gather -> FFN -> branch scatter (weights async-staged in LDS)
    ffn_kernel<<<N_ROWS / TILE_M, 256, 0, stream>>>(x, gf, b1, b2, wfrag, idxs, pidx, out);
}
